// LMUFFT_7499012899105
// MI455X (gfx1250) — compile-verified, matrix operand order checked by reading, not run on hardware
//
#include <hip/hip_runtime.h>
#include <hip/hip_bf16.h>

// ---------------------------------------------------------------------------
// LMU layer for MI455X (gfx1250, wave32, WMMA).
// Causal FFT-convolution re-expressed as triangular-Toeplitz GEMM running on
// v_wmma_f32_16x16x32_f16 with f32 accumulation.
//  - B/W fragments: two aligned global b128 loads per fragment (ISA layout
//    puts 8 contiguous K values per lane per 4-VGPR half).
//  - Toeplitz A fragment: u prefix staged to LDS once per block (async-to-LDS
//    when available), fragment built with ds_load_u16 on the DS pipe.
// ---------------------------------------------------------------------------

typedef _Float16 f16x16 __attribute__((ext_vector_type(16)));
typedef _Float16 f16x8  __attribute__((ext_vector_type(8)));
typedef float    f32x8  __attribute__((ext_vector_type(8)));
typedef int      i32x4  __attribute__((vector_size(16)));

#define BATCH   8
#define SEQ     4096
#define DIN     512
#define DHID    1024
#define KMEM    256
#define KCAT    (KMEM + DIN)     // 768
#define UPAD    64               // zero pad for negative Toeplitz taps
#define USTRIDE (SEQ + UPAD)     // per-batch stride of padded u (16B aligned)

#if __has_builtin(__builtin_amdgcn_global_load_async_to_lds_b128)
#define HAVE_ASYNC_LDS 1
#endif

// K-index of WMMA 16-bit A/B fragment element pair held in VGPR v by lane
// half g (per CDNA5 ISA 7.12.2: 16-bit A-matrix 16x32 layout).
__device__ __forceinline__ int frag_kb(int v, int g) {
    return ((v & 4) << 2) + (g << 3) + ((v & 3) << 1);
}

__device__ __forceinline__ f16x16 cat8(f16x8 lo, f16x8 hi) {
    return __builtin_shufflevector(lo, hi, 0, 1, 2, 3, 4, 5, 6, 7,
                                   8, 9, 10, 11, 12, 13, 14, 15);
}

// ---------------- fp32 -> f16 conversion (grid-stride) ----------------------
__global__ void cvt_f32_to_f16(const float* __restrict__ in,
                               _Float16* __restrict__ out, int n) {
    int i = blockIdx.x * blockDim.x + threadIdx.x;
    int stride = gridDim.x * blockDim.x;
    for (; i < n; i += stride) out[i] = (_Float16)in[i];
}

// ---------------- u = relu(x . W_u + b), one wave per row -------------------
__global__ void u_kernel(const float* __restrict__ x,
                         const float* __restrict__ wu,
                         const float* __restrict__ bu,
                         _Float16* __restrict__ up) {
    int wid  = (blockIdx.x * blockDim.x + threadIdx.x) >> 5;  // row 0..B*S-1
    int lane = threadIdx.x & 31;
    int b = wid >> 12;           // / SEQ
    int s = wid & (SEQ - 1);
    const float* xr = x + (size_t)wid * DIN;
    float acc = 0.f;
    #pragma unroll 4
    for (int i = lane; i < DIN; i += 32) acc += xr[i] * wu[i];
    #pragma unroll
    for (int off = 16; off > 0; off >>= 1) acc += __shfl_down(acc, off, 32);
    if (lane == 0) {
        float v = acc + bu[0];
        v = v > 0.f ? v : 0.f;
        _Float16* ub = up + (size_t)b * USTRIDE;
        ub[UPAD + s] = (_Float16)v;
        if (s < UPAD) ub[s] = (_Float16)0.f;   // zero the causal pad
    }
}

// ---------------- causal conv as Toeplitz GEMM on WMMA ----------------------
// m[b,t,k] = sum_{d=0..t} u[b,t-d] * H[k,d]
// Block: one batch x one 16-row t-tile, 8 waves cover all 256 k columns.
__global__ void __launch_bounds__(256)
lmu_conv_wmma(const _Float16* __restrict__ up,  // (B,USTRIDE)
              const _Float16* __restrict__ Hf,  // (KMEM,SEQ)
              _Float16* __restrict__ mf) {      // (B*SEQ,KMEM)
    __shared__ _Float16 uS[UPAD + SEQ + 64];    // staged padded u prefix

    int blk  = blockIdx.x;
    int b    = blk >> 8;                 // / (SEQ/16)
    int t0   = (blk & 255) << 4;
    int lane = threadIdx.x & 31;
    int w    = threadIdx.x >> 5;         // wave 0..7
    int m    = lane & 15;                // A-row / B-col within tile
    int g    = lane >> 4;                // lane-half selects K sub-block
    int k0   = w << 5;                   // this wave's first output column

    // ---- stage u[b, 0 .. t0+15] (plus causal zero pad) into LDS ----
    const _Float16* ugsrc = up + (size_t)b * USTRIDE;
    int chunks = (UPAD + t0 + 16 + 7) >> 3;          // 16B chunks
    for (int c = threadIdx.x; c < chunks; c += 256) {
#if defined(HAVE_ASYNC_LDS)
        __builtin_amdgcn_global_load_async_to_lds_b128(
            (__attribute__((address_space(1))) i32x4*)(ugsrc + c * 8),
            (__attribute__((address_space(3))) i32x4*)(&uS[c * 8]), 0, 0);
#else
        *(f16x8*)&uS[c * 8] = *(const f16x8*)(ugsrc + c * 8);
#endif
    }
#if defined(HAVE_ASYNC_LDS)
#if __has_builtin(__builtin_amdgcn_s_wait_asynccnt)
    __builtin_amdgcn_s_wait_asynccnt(0);
#else
    asm volatile("s_wait_asynccnt 0x0" ::: "memory");
#endif
#endif
    __syncthreads();

    const f16x8* Hr0 = (const f16x8*)(Hf + (size_t)(k0 + m) * SEQ);
    const f16x8* Hr1 = (const f16x8*)(Hf + (size_t)(k0 + 16 + m) * SEQ);

    f32x8 c0 = {};
    f32x8 c1 = {};

    // Causal K loop: only d-tiles with d0 <= t (skips ~half the work).
    for (int d0 = 0; d0 <= t0 + 15; d0 += 32) {
        // A fragment: Toeplitz taps gathered from LDS (DS pipe).
        f16x16 a;
        int baseL = UPAD + t0 + m - d0;
        #pragma unroll
        for (int v = 0; v < 8; ++v) {
            int kb = frag_kb(v, g);
            a[2 * v]     = uS[baseL - kb];       // >= 18, staged zeros cover
            a[2 * v + 1] = uS[baseL - kb - 1];
        }
        // B fragments: two aligned b128 loads each.
        int q = (d0 >> 3) + g;
        f16x16 b0 = cat8(Hr0[q], Hr0[q + 2]);
        f16x16 b1 = cat8(Hr1[q], Hr1[q + 2]);

        c0 = __builtin_amdgcn_wmma_f32_16x16x32_f16(false, a, false, b0,
                                                    (short)0, c0, false, false);
        c1 = __builtin_amdgcn_wmma_f32_16x16x32_f16(false, a, false, b1,
                                                    (short)0, c1, false, false);
    }

    #pragma unroll
    for (int r = 0; r < 8; ++r) {
        int t = t0 + r + (g << 3);       // C/D layout: VGPR r -> row r (+8)
        _Float16* mrow = mf + (size_t)(b * SEQ + t) * KMEM;
        mrow[k0 + m]      = (_Float16)c0[r];
        mrow[k0 + 16 + m] = (_Float16)c1[r];
    }
}

// ---------------- h = relu([m,x] . W_h^T + b) on WMMA -----------------------
// Block: 16 rows x 1024 cols; 8 waves x 128 cols (8 accumulator tiles each).
__global__ void __launch_bounds__(256)
out_gemm_wmma(const _Float16* __restrict__ mf,  // (R,256)
              const _Float16* __restrict__ xf,  // (R,512)
              const _Float16* __restrict__ Wf,  // (1024,768)
              const float* __restrict__ bias,   // (1024)
              float* __restrict__ out) {
    int r0   = blockIdx.x << 4;
    int lane = threadIdx.x & 31;
    int w    = threadIdx.x >> 5;
    int m    = lane & 15;
    int g    = lane >> 4;
    int n0   = w << 7;                   // 128 cols per wave

    f32x8 acc[8];
    #pragma unroll
    for (int t = 0; t < 8; ++t) acc[t] = (f32x8){};

    const _Float16* mrow = mf + (size_t)(r0 + m) * KMEM;
    const _Float16* xrow = xf + (size_t)(r0 + m) * DIN;

    for (int kq = 0; kq < KCAT; kq += 32) {      // uniform; no divergence
        // A fragment: two aligned b128 loads from m (k<256) or x (k>=256).
        const f16x8* arow = (kq < KMEM)
            ? (const f16x8*)(mrow + kq)
            : (const f16x8*)(xrow + (kq - KMEM));
        f16x16 a = cat8(arow[g], arow[g + 2]);

        #pragma unroll
        for (int t = 0; t < 8; ++t) {
            int col = n0 + (t << 4) + m;
            const f16x8* wrow = (const f16x8*)(Wf + (size_t)col * KCAT + kq);
            f16x16 bfr = cat8(wrow[g], wrow[g + 2]);
            acc[t] = __builtin_amdgcn_wmma_f32_16x16x32_f16(
                false, a, false, bfr, (short)0, acc[t], false, false);
        }
    }

    const int R = BATCH * SEQ;
    #pragma unroll
    for (int t = 0; t < 8; ++t) {
        int col = n0 + (t << 4) + m;
        float bv = bias[col];
        #pragma unroll
        for (int r = 0; r < 8; ++r) {
            int row = r0 + r + (g << 3);
            float v = acc[t][r] + bv;
            v = v > 0.f ? v : 0.f;
            out[(size_t)row * DHID + col] = v;
            if ((row & (SEQ - 1)) == SEQ - 1)   // h[:, -1, :] second output
                out[(size_t)R * DHID + (size_t)(row >> 12) * DHID + col] = v;
        }
    }
}

// ---------------------------------------------------------------------------
extern "C" void kernel_launch(void* const* d_in, const int* in_sizes, int n_in,
                              void* d_out, int out_size, void* d_ws,
                              size_t ws_size, hipStream_t stream) {
    const float* x   = (const float*)d_in[0];   // (8,4096,512)
    const float* Wuw = (const float*)d_in[1];   // (1,512)
    const float* Wub = (const float*)d_in[2];   // (1,)
    const float* Whw = (const float*)d_in[3];   // (1024,768)
    const float* Whb = (const float*)d_in[4];   // (1024,)
    const float* H   = (const float*)d_in[5];   // (256,4096)
    float* out = (float*)d_out;

    char* ws = (char*)d_ws;
    size_t off = 0;
    auto carve = [&](size_t bytes) -> void* {
        void* p = ws + off;
        off = (off + bytes + 255) & ~(size_t)255;
        return p;
    };
    _Float16* Hf = (_Float16*)carve((size_t)KMEM * SEQ * 2);          //  2 MB
    _Float16* xf = (_Float16*)carve((size_t)BATCH * SEQ * DIN * 2);   // 34 MB
    _Float16* Wf = (_Float16*)carve((size_t)DHID * KCAT * 2);         // 1.5MB
    _Float16* up = (_Float16*)carve((size_t)BATCH * USTRIDE * 2);     // 65 KB
    _Float16* mf = (_Float16*)carve((size_t)BATCH * SEQ * KMEM * 2);  // 17 MB

    cvt_f32_to_f16<<<2048, 256, 0, stream>>>(H, Hf, KMEM * SEQ);
    cvt_f32_to_f16<<<4096, 256, 0, stream>>>(x, xf, BATCH * SEQ * DIN);
    cvt_f32_to_f16<<<1024, 256, 0, stream>>>(Whw, Wf, DHID * KCAT);

    u_kernel<<<(BATCH * SEQ) / 8, 256, 0, stream>>>(x, Wuw, Wub, up);

    lmu_conv_wmma<<<BATCH * (SEQ / 16), 256, 0, stream>>>(up, Hf, mf);

    out_gemm_wmma<<<(BATCH * SEQ) / 16, 256, 0, stream>>>(mf, xf, Wf, Whb, out);
}
